// RotationConditionerBlock_33380485825345
// MI455X (gfx1250) — compile-verified, matrix operand order checked by reading, not run on hardware
//
#include <hip/hip_runtime.h>
#include <hip/hip_bf16.h>

// ---------------------------------------------------------------------------
// Problem constants (from the reference)
// ---------------------------------------------------------------------------
#define SEQ   2048
#define ND    128
#define NA    64
#define NHEAD 8
#define CH    64
#define TH3   24          // 3 * NHEAD tripled heads
#define OKQ   1024        // H*C*2
#define OV    3072        // ND*H*3

typedef __attribute__((ext_vector_type(16))) __bf16 v16bf;
typedef __attribute__((ext_vector_type(2)))  __bf16 v2bf;
typedef __attribute__((ext_vector_type(8)))  float  v8f;

union Frag {
    v16bf v;
    unsigned int d[8];
    unsigned short h[16];
};

__device__ __forceinline__ unsigned short f2bf(float f) {
    unsigned u = __builtin_bit_cast(unsigned, f);
    unsigned r = u + 0x7FFFu + ((u >> 16) & 1u);   // round-to-nearest-even
    return (unsigned short)(r >> 16);
}

// Pack two f32 -> packed bf16 dword (hardware cvt_pk when available).
__device__ __forceinline__ unsigned pack_bf16(float lo, float hi) {
#if __has_builtin(__builtin_amdgcn_cvt_pk_bf16_f32)
    v2bf p = __builtin_amdgcn_cvt_pk_bf16_f32(lo, hi);
    return __builtin_bit_cast(unsigned, p);
#else
    union { v2bf b; unsigned u; } x;
    x.b[0] = (__bf16)lo;
    x.b[1] = (__bf16)hi;
    return x.u;
#endif
}

__device__ __forceinline__ v8f zero8() {
    v8f z;
#pragma unroll
    for (int i = 0; i < 8; ++i) z[i] = 0.0f;
    return z;
}

// Per-VGPR dword delta inside a 16x32 bf16 fragment (k/2 minus the lane part).
__device__ __forceinline__ int frag_dv(int v) {
    return (v >> 2) * 8 + (v & 3);
}

// ---------------------------------------------------------------------------
// Projection: K,Q for the 3 groups (nodes / pos / aux).  Small FLOPs -> VALU.
// Output: Kbf/Qbf bf16, layout [24][SEQ][64] (head-major, row-major in c).
// ---------------------------------------------------------------------------
__global__ void proj_kq_kernel(const float* __restrict__ nodes,
                               const float* __restrict__ pos,
                               const float* __restrict__ aux,
                               const float* __restrict__ Wn, const float* __restrict__ bn,
                               const float* __restrict__ Wp, const float* __restrict__ bp,
                               const float* __restrict__ Wa,
                               unsigned short* __restrict__ Kbf,
                               unsigned short* __restrict__ Qbf) {
    unsigned tid = blockIdx.x * blockDim.x + threadIdx.x;
    unsigned s = tid & (SEQ - 1);
    unsigned o = (tid >> 11) & (OKQ - 1);
    unsigned g = tid >> 21;
    if (g >= 3) return;

    float acc = 0.0f;
    if (g == 0) {
        acc = bn[o];
        const float* x = nodes + s * ND;
        const float* w = Wn + o * ND;
#pragma unroll 8
        for (int d = 0; d < ND; ++d) acc += x[d] * w[d];
    } else if (g == 1) {
        acc = bp[o];
        const float* x = pos + s * 6;
        const float* w = Wp + o * 6;
#pragma unroll
        for (int d = 0; d < 6; ++d) acc += x[d] * w[d];
    } else {
        const float* x = aux + s * NA;
        const float* w = Wa + o * NA;
#pragma unroll 8
        for (int d = 0; d < NA; ++d) acc += x[d] * w[d];
    }
    unsigned hh = o >> 7;
    unsigned c  = o & 127;
    unsigned ht = g * NHEAD + hh;
    unsigned short b = f2bf(acc);
    if (c < CH) Kbf[(ht * SEQ + s) * CH + c]        = b;
    else        Qbf[(ht * SEQ + s) * CH + (c - CH)] = b;
}

// ---------------------------------------------------------------------------
// Projection: V = nodes @ Wv^T + bv, stored TRANSPOSED as bf16:
//   VbfT[ht][d][j] , layout [24][128][2048]
// ---------------------------------------------------------------------------
__global__ void proj_v_kernel(const float* __restrict__ nodes,
                              const float* __restrict__ Wv,
                              const float* __restrict__ bv,
                              unsigned short* __restrict__ VbfT) {
    unsigned tid = blockIdx.x * blockDim.x + threadIdx.x;
    unsigned s = tid & (SEQ - 1);
    unsigned o = tid >> 11;
    if (o >= OV) return;

    float acc = bv[o];
    const float* x = nodes + s * ND;
    const float* w = Wv + o * ND;
#pragma unroll 8
    for (int d = 0; d < ND; ++d) acc += x[d] * w[d];

    unsigned ht = o >> 7;
    unsigned d2 = o & 127;
    VbfT[(ht * (unsigned)ND + d2) * (unsigned)SEQ + s] = f2bf(acc);
}

// ---------------------------------------------------------------------------
// Flash attention, TRANSPOSED-GEMM formulation (no LDS layout shuffle):
//
//   S'[j][i] = q_j . k_i     : A = Q j-rows (streamed), B = K i-rows (fixed)
//   O'[d][i] = sum_j V^T P'  : A = V^T d-rows (streamed), B = P' (lane-local!)
//
// C/D layout (16x16 f32): lane L -> col n = L&15, VGPR r -> row m=(L>>4)*8+r.
// With scores transposed, lane L owns output row i = L&15 and 16 of the 32
// j-scores of the chunk (partner L^16 owns the rest):
//   * row max / row sum  = local reduce + one shfl_xor(16)
//   * alpha, 1/l         = per-lane scalars (no broadcasts)
//   * P B-fragment k-map = exactly the j-values this lane holds -> pure
//     register packing, zero LDS traffic in the hot loop.
// ---------------------------------------------------------------------------
__global__ __launch_bounds__(768, 1)
void attn_kernel(const unsigned short* __restrict__ Kbf,
                 const unsigned short* __restrict__ Qbf,
                 const unsigned short* __restrict__ VbfT,
                 float* __restrict__ out) {
    __shared__ float red[16][ND];                 // cross-head reduction buffer

    const int lane = threadIdx.x;
    const int wid  = threadIdx.y;                 // tripled head 0..23
    const int i0   = blockIdx.x * 16;
    const float scale = 0.35355339059327373f;     // 1/sqrt(num_heads=8)

    const unsigned int* Kp = (const unsigned int*)Kbf;
    const unsigned int* Qp = (const unsigned int*)Qbf;
    const unsigned int* Vp = (const unsigned int*)VbfT;

    const int lm = lane & 15;    // B-role col (output row i) / A-role row
    const int lh = lane >> 4;    // half select

    // B fragments of K rows [i0..i0+15] (fixed for the block), c in 2 chunks.
    Frag bK[2];
    {
        unsigned kInv = ((unsigned)wid * SEQ + (unsigned)(i0 + lm)) * (CH / 2) + (unsigned)(lh * 4);
#pragma unroll
        for (int cc = 0; cc < 2; ++cc)
#pragma unroll
            for (int v = 0; v < 8; ++v)
                bK[cc].d[v] = Kp[kInv + cc * 16 + frag_dv(v)];
    }

    // Per-lane 32-bit DWORD-offset invariants (hoisted out of the j-loop).
    unsigned qInv = ((unsigned)wid * SEQ + (unsigned)lm) * (CH / 2) + (unsigned)(lh * 4);
    unsigned vInv[8];
#pragma unroll
    for (int t = 0; t < 8; ++t)
        vInv[t] = ((unsigned)wid * ND + (unsigned)(t * 16 + lm)) * (SEQ / 2) + (unsigned)(lh * 4);

    v8f acc[8];                  // O'[d][i]: acc[t][r] = O[i0+lm][t*16+lh*8+r]
#pragma unroll
    for (int t = 0; t < 8; ++t) acc[t] = zero8();
    float rm = -1e30f;           // running max of row i = lm
    float rl = 0.0f;             // running sum of row i = lm

    for (int j0 = 0; j0 < SEQ; j0 += 32) {
        // ---- S' = Q(A, streamed) x K(B, fixed): two 16-j tiles, 2 WMMAs each
        v8f sfr[2];
#pragma unroll
        for (int jt = 0; jt < 2; ++jt) {
            unsigned qb = qInv + (unsigned)((j0 + jt * 16) * (CH / 2));
            Frag aQ0, aQ1;
#pragma unroll
            for (int v = 0; v < 8; ++v) aQ0.d[v] = Qp[qb + frag_dv(v)];
#pragma unroll
            for (int v = 0; v < 8; ++v) aQ1.d[v] = Qp[qb + 16 + frag_dv(v)];
            v8f sacc = zero8();
            sacc = __builtin_amdgcn_wmma_f32_16x16x32_bf16(
                       false, aQ0.v, false, bK[0].v, (short)0, sacc, false, false);
            sacc = __builtin_amdgcn_wmma_f32_16x16x32_bf16(
                       false, aQ1.v, false, bK[1].v, (short)0, sacc, false, false);
            sfr[jt] = sacc;
        }

        // ---- per-lane online softmax (row i = lm; 16 j-values + partner) ----
        float sc[16];
#pragma unroll
        for (int jt = 0; jt < 2; ++jt)
#pragma unroll
            for (int r = 0; r < 8; ++r) sc[jt * 8 + r] = sfr[jt][r] * scale;

        float mloc = sc[0];
#pragma unroll
        for (int e = 1; e < 16; ++e) mloc = fmaxf(mloc, sc[e]);
        mloc = fmaxf(mloc, __shfl_xor(mloc, 16, 32));   // partner's 16 j-values
        float mnew  = fmaxf(rm, mloc);
        float alpha = __expf(rm - mnew);
        rm = mnew;

        // P' B-fragment is lane-local: k-map hits exactly this lane's j-values.
        Frag bP;
        float ssum = 0.0f;
#pragma unroll
        for (int v = 0; v < 8; ++v) {
            int g  = v >> 2;          // score fragment (j tile)
            int i3 = v & 3;           // pair index -> r = 2*i3, 2*i3+1
            float p0 = __expf(sc[g * 8 + 2 * i3]     - mnew);
            float p1 = __expf(sc[g * 8 + 2 * i3 + 1] - mnew);
            ssum += p0 + p1;
            bP.d[v] = pack_bf16(p0, p1);
        }
        ssum += __shfl_xor(ssum, 16, 32);
        rl = rl * alpha + ssum;

        // ---- rescale accumulators (scalar per lane; skip once max settles) ----
        if (__ballot(alpha != 1.0f)) {
#pragma unroll
            for (int t = 0; t < 8; ++t)
#pragma unroll
                for (int r = 0; r < 8; ++r) acc[t][r] *= alpha;
        }

        // ---- O' += V^T(A, streamed) x P'(B): 8 d-tiles ----
        unsigned vj = (unsigned)(j0 >> 1);
#pragma unroll
        for (int t = 0; t < 8; ++t) {
            Frag aV;
#pragma unroll
            for (int v = 0; v < 8; ++v) aV.d[v] = Vp[vInv[t] + vj + frag_dv(v)];
            acc[t] = __builtin_amdgcn_wmma_f32_16x16x32_bf16(
                         false, aV.v, false, bP.v, (short)0, acc[t], false, false);
        }
    }

    // ---- normalize (per-lane scalar) ----
    float rinv = 1.0f / rl;
#pragma unroll
    for (int t = 0; t < 8; ++t)
#pragma unroll
        for (int r = 0; r < 8; ++r) acc[t][r] *= rinv;

    // ---- deterministic fixed-order reduction across the 24 heads ----
    for (int w = 0; w < TH3; ++w) {
        if (wid == w) {
#pragma unroll
            for (int t = 0; t < 8; ++t)
#pragma unroll
                for (int r = 0; r < 8; ++r) {
                    int d = t * 16 + lh * 8 + r;
                    if (w == 0) red[lm][d]  = acc[t][r];
                    else        red[lm][d] += acc[t][r];
                }
        }
        __syncthreads();
    }
    if (wid == 0) {
#pragma unroll
        for (int t = 0; t < 8; ++t)
#pragma unroll
            for (int r = 0; r < 8; ++r) {
                int d = t * 16 + lh * 8 + r;
                out[(i0 + lm) * ND + d] = red[lm][d];
            }
    }
}

// ---------------------------------------------------------------------------
// Launch
// ---------------------------------------------------------------------------
extern "C" void kernel_launch(void* const* d_in, const int* in_sizes, int n_in,
                              void* d_out, int out_size, void* d_ws, size_t ws_size,
                              hipStream_t stream) {
    const float* nodes = (const float*)d_in[0];
    const float* pos   = (const float*)d_in[1];
    const float* aux   = (const float*)d_in[2];
    const float* Wn    = (const float*)d_in[3];
    const float* bn    = (const float*)d_in[4];
    const float* Wp    = (const float*)d_in[5];
    const float* bp    = (const float*)d_in[6];
    const float* Wa    = (const float*)d_in[7];
    const float* Wv    = (const float*)d_in[8];
    const float* bv    = (const float*)d_in[9];
    float* out = (float*)d_out;

    // Workspace layout (bytes):
    //   Kbf  [24][2048][64]  bf16 :  6,291,456
    //   Qbf  [24][2048][64]  bf16 :  6,291,456
    //   VbfT [24][128][2048] bf16 : 12,582,912
    char* ws = (char*)d_ws;
    unsigned short* Kbf  = (unsigned short*)(ws);
    unsigned short* Qbf  = (unsigned short*)(ws + (size_t)TH3 * SEQ * CH * 2);
    unsigned short* VbfT = (unsigned short*)(ws + (size_t)TH3 * SEQ * CH * 4);

    {
        unsigned total = 3u << 21;                  // 3 * 1024 * 2048
        proj_kq_kernel<<<total / 256, 256, 0, stream>>>(
            nodes, pos, aux, Wn, bn, Wp, bp, Wa, Kbf, Qbf);
    }
    {
        unsigned total = (unsigned)OV * SEQ;        // 3072 * 2048
        proj_v_kernel<<<total / 256, 256, 0, stream>>>(nodes, Wv, bv, VbfT);
    }

    dim3 blk(32, TH3, 1);
    dim3 grd(SEQ / 16, 1, 1);
    attn_kernel<<<grd, blk, 0, stream>>>(Kbf, Qbf, VbfT, out);
}
